// MoShead_52063593562354
// MI455X (gfx1250) — matrix operand order repaired and checked by reading, not compiled, not run on hardware
//
#include <hip/hip_runtime.h>
#include <cstdint>
#include <cstddef>

// ---------------------------------------------------------------------------
// MoS head for MI455X (gfx1250, wave32, WMMA + TDM).
// Dominant op: (5120 x 1024) @ (1024 x 32000) bf16 WMMA, computed twice
// (sum-exp pass + probability pass) instead of materializing 655MB of logits.
// W_dec in bf16 (65MB) stays L2-resident (192MB), so the recompute is ~free in
// HBM traffic. GEMM operands pre-swizzled into native WMMA fragment layouts.
// GEMM2 A-tiles are DMA'd into LDS by the Tensor Data Mover, double-buffered,
// synchronized with s_wait_tensorcnt + workgroup barriers.
// ---------------------------------------------------------------------------

typedef __attribute__((ext_vector_type(16))) __bf16 v16bf;
typedef __attribute__((ext_vector_type(8)))  float  v8f;

#define NTOKENS 512     // 64*8 rows of x
#define NHID    1024
#define NEXP    10
#define NINPD   1024
#define NTOK    32000

#if defined(__has_builtin)
#if __has_builtin(__builtin_amdgcn_tensor_load_to_lds)
#define HAVE_TDM 1
#else
#define HAVE_TDM 0
#endif
#if __has_builtin(__builtin_amdgcn_s_wait_tensorcnt)
#define HAVE_WTC 1
#else
#define HAVE_WTC 0
#endif
#else
#define HAVE_TDM 0
#define HAVE_WTC 0
#endif

// ---- compile-time probes (visible in stderr even on success) --------------
#if HAVE_TDM
#warning CDNA5_PROBE: tensor_load_to_lds builtin AVAILABLE - TDM path enabled
#else
#warning CDNA5_PROBE: tensor_load_to_lds builtin MISSING - cooperative-copy fallback in use
#endif
#if HAVE_WTC
#warning CDNA5_PROBE: s_wait_tensorcnt builtin AVAILABLE
#else
#warning CDNA5_PROBE: s_wait_tensorcnt builtin MISSING - inline asm fallback
#endif
#if defined(__clang_major__) && (__clang_major__ >= 23)
#warning CDNA5_PROBE: clang >= 23 - using 6-arg tensor_load_to_lds form
#else
#warning CDNA5_PROBE: clang <= 22 - using 5-arg tensor_load_to_lds form
#endif

struct Q8 { unsigned int u[8]; };   // one lane's 32-byte fragment chunk

__device__ __forceinline__ v16bf ld_frag(const Q8* __restrict__ p) {
    Q8 q = *p;
    return __builtin_bit_cast(v16bf, q);
}

__device__ __forceinline__ unsigned short f32_to_bf16(float f) {
    unsigned int u = __float_as_uint(f);
    u += 0x7FFFu + ((u >> 16) & 1u);          // round to nearest even
    return (unsigned short)(u >> 16);
}

__device__ __forceinline__ void wait_tensorcnt_le1() {
#if HAVE_WTC
    __builtin_amdgcn_s_wait_tensorcnt(1);
#else
    asm volatile("s_wait_tensorcnt 0x1" ::: "memory");
#endif
}
__device__ __forceinline__ void wait_tensorcnt_le0() {
#if HAVE_WTC
    __builtin_amdgcn_s_wait_tensorcnt(0);
#else
    asm volatile("s_wait_tensorcnt 0x0" ::: "memory");
#endif
}

// ---------------------------------------------------------------------------
// TDM: DMA `nrows` A-fragments (1024B each, spaced 32KB apart) into LDS.
// D# group0: count=1, lds_addr, global_addr[56:0], type=2.
// D# group1: data_size=4B, tensor_dim0=256, tensor_dim1=10240,
//            tile_dim0=256, tile_dim1=nrows, tensor_dim0_stride=8192 elems.
// ---------------------------------------------------------------------------
#if HAVE_TDM
typedef unsigned uv4 __attribute__((ext_vector_type(4)));
typedef int      iv4 __attribute__((ext_vector_type(4)));
typedef int      iv8 __attribute__((ext_vector_type(8)));

__device__ __forceinline__ void tdm_load_frags(const void* gbase, unsigned ldsOff,
                                               int nrows) {
    const unsigned long long ga = (unsigned long long)(uintptr_t)gbase;
    uv4 g0;
    g0[0] = 1u;                                               // count=1 (user D#)
    g0[1] = ldsOff;                                           // lds_addr (bytes)
    g0[2] = (unsigned)ga;                                     // global_addr[95:64]
    g0[3] = (unsigned)((ga >> 32) & 0x01FFFFFFu) | (2u << 30);// addr[56:32]|type=2
    iv8 g1;
    g1[0] = (int)(2u << 16);                                  // data_size=4B
    g1[1] = (int)((256u & 0xFFFFu) << 16);                    // tensor_dim0 lo16
    g1[2] = (int)(((256u >> 16) & 0xFFFFu) |                  // tensor_dim0 hi16
                  ((10240u & 0xFFFFu) << 16));                // tensor_dim1 lo16
    g1[3] = (int)(((10240u >> 16) & 0xFFFFu) |                // tensor_dim1 hi16
                  (256u << 16));                              // tile_dim0=256
    g1[4] = (int)(nrows & 0xFFFF);                            // tile_dim1
    g1[5] = (int)8192u;                                       // dim0_stride lo32
    g1[6] = 0;
    g1[7] = 0;
    iv4 z4 = {0, 0, 0, 0};
#if defined(__clang_major__) && (__clang_major__ >= 23)
    iv8 z8 = {0, 0, 0, 0, 0, 0, 0, 0};
    __builtin_amdgcn_tensor_load_to_lds(g0, g1, z4, z4, z8, 0);
#else
    __builtin_amdgcn_tensor_load_to_lds(g0, g1, z4, z4, 0);
#endif
}
#endif

// ---------------------------------------------------------------------------
// Fragment-layout converters (fp32 -> bf16, swizzled to WMMA register order).
// ---------------------------------------------------------------------------
__global__ void convert_B_frags(const float* __restrict__ W,
                                unsigned short* __restrict__ out, int ncols) {
    const int nt = blockIdx.x, kb = blockIdx.y;
    const int t  = threadIdx.x;
    const int kl = t >> 4, nl = t & 15;
    const float f = W[(size_t)(kb * 32 + kl) * ncols + nt * 16 + nl];
    const int lane = nl + (((kl >> 3) & 1) << 4);
    const int h    = (kl & 7) | ((kl >> 4) << 3);
    out[((size_t)(nt * 32 + kb)) * 512 + lane * 16 + h] = f32_to_bf16(f);
}

__global__ void convert_x_frags(const float* __restrict__ x,
                                unsigned short* __restrict__ out) {
    const int mt = blockIdx.x, kb = blockIdx.y;
    const int t  = threadIdx.x;
    const int kl = t & 31, m = t >> 5;
    const float f = x[(size_t)(mt * 16 + m) * NHID + kb * 32 + kl];
    const int lane = m + (((kl >> 3) & 1) << 4);
    const int v    = ((kl >> 4) << 2) | ((kl >> 1) & 3);
    const int h    = (v << 1) | (kl & 1);
    out[((size_t)(mt * 32 + kb)) * 512 + lane * 16 + h] = f32_to_bf16(f);
}

// ---------------------------------------------------------------------------
// mulogvar = x @ W_reduce + b_reduce ; mu/logvar outputs ; stick-breaking pis
// ---------------------------------------------------------------------------
__global__ void reduce_pis(const float* __restrict__ x,
                           const float* __restrict__ Wr,
                           const float* __restrict__ br,
                           float* __restrict__ out_mu,
                           float* __restrict__ out_lv,
                           float* __restrict__ pis) {
    const int n = blockIdx.x;
    const int lane = threadIdx.x;        // 32 threads (one wave)
    float acc[18];
#pragma unroll
    for (int j = 0; j < 18; ++j) acc[j] = 0.f;
    for (int i = lane; i < NHID; i += 32) {
        const float xv = x[(size_t)n * NHID + i];
#pragma unroll
        for (int j = 0; j < 18; ++j) acc[j] += xv * Wr[i * 18 + j];
    }
#pragma unroll
    for (int j = 0; j < 18; ++j) {
        float s = acc[j];
        s += __shfl_xor(s, 16, 32);
        s += __shfl_xor(s,  8, 32);
        s += __shfl_xor(s,  4, 32);
        s += __shfl_xor(s,  2, 32);
        s += __shfl_xor(s,  1, 32);
        acc[j] = s;
    }
    if (lane == 0) {
        float vs[10];
        for (int j = 0; j < 9; ++j) {
            const float m = acc[j] + br[j];
            out_mu[n * 9 + j] = m;
            vs[j] = 1.f / (1.f + __expf(-m));
        }
        for (int j = 0; j < 9; ++j) out_lv[n * 9 + j] = acc[9 + j] + br[9 + j];
        vs[9] = 1.f;
        float cum = 1.f;
        for (int k = 0; k < 10; ++k) {
            if (k) cum *= (1.f - vs[k - 1]);
            pis[n * 10 + k] = cum * vs[k];
        }
    }
}

__global__ void zero_f32(float* p, int n) {
    const int i = blockIdx.x * blockDim.x + threadIdx.x;
    if (i < n) p[i] = 0.f;
}

// ---------------------------------------------------------------------------
// Direct-from-global WMMA mainloop (used by the small GEMM1).
// ---------------------------------------------------------------------------
template <int NRT>
__device__ __forceinline__ void wmma_mainloop(const Q8* __restrict__ A,
                                              const Q8* __restrict__ B,
                                              int mtile0, int ctile,
                                              v8f acc[NRT]) {
    const int lane = threadIdx.x & 31;
#pragma unroll
    for (int i = 0; i < NRT; ++i) {
        v8f z = {0.f, 0.f, 0.f, 0.f, 0.f, 0.f, 0.f, 0.f};
        acc[i] = z;
    }
    for (int kb = 0; kb < 32; ++kb) {
        const v16bf bv = ld_frag(B + ((size_t)ctile * 32 + kb) * 32 + lane);
        if (kb + 1 < 32)
            __builtin_prefetch((const void*)(B + ((size_t)ctile * 32 + kb + 1) * 32 + lane), 0, 3);
#pragma unroll
        for (int rt = 0; rt < NRT; ++rt) {
            const v16bf av = ld_frag(A + ((size_t)(mtile0 + rt) * 32 + kb) * 32 + lane);
            acc[rt] = __builtin_amdgcn_wmma_f32_16x16x32_bf16(
                false, av, false, bv, (short)0, acc[rt], false, false);
        }
    }
}

// ---------------------------------------------------------------------------
// TDM-staged WMMA mainloop for GEMM2 (NRT=10 row-tiles, 160-row workgroup
// tile). Wave 0 DMAs 10KB of shared A-fragments per K-step into a
// double-buffered LDS staging area; all waves read fragments via ds_load_b128.
// All 10 fragments are preloaded to registers so the 10 WMMAs issue as a
// burst without interleaved LDS waits.
// ---------------------------------------------------------------------------
__device__ __forceinline__ void wmma_mainloop_lds10(const Q8* __restrict__ A,
                                                    const Q8* __restrict__ B,
                                                    int mtile0, int ctile,
                                                    v8f acc[10], char* smemA) {
    const int wave = threadIdx.x >> 5;
    const int lane = threadIdx.x & 31;
#pragma unroll
    for (int i = 0; i < 10; ++i) {
        v8f z = {0.f, 0.f, 0.f, 0.f, 0.f, 0.f, 0.f, 0.f};
        acc[i] = z;
    }
#if HAVE_TDM
    if (wave == 0)
        tdm_load_frags((const char*)A + (size_t)(mtile0 * 32) * 1024,
                       (unsigned)(uintptr_t)smemA, 10);
#endif
    for (int kb = 0; kb < 32; ++kb) {
        char* cur = smemA + (kb & 1) * 10240;
#if HAVE_TDM
        if (wave == 0) {
            if (kb < 31) {
                tdm_load_frags((const char*)A + (size_t)(mtile0 * 32 + kb + 1) * 1024,
                               (unsigned)(uintptr_t)(smemA + ((kb + 1) & 1) * 10240), 10);
                wait_tensorcnt_le1();       // oldest (current buffer) complete
            } else {
                wait_tensorcnt_le0();
            }
        }
        __syncthreads();                    // staged tile visible to all waves
#else
        {   // fallback: cooperative synchronous copy of the 10 fragments
            uint4* ldst = (uint4*)cur;
            for (int i = threadIdx.x; i < 640; i += 256) {
                const int rt = i >> 6, off = i & 63;
                ldst[i] = ((const uint4*)((const char*)A +
                          ((size_t)((mtile0 + rt) * 32 + kb)) * 1024))[off];
            }
        }
        __syncthreads();
#endif
        const v16bf bv = ld_frag(B + ((size_t)ctile * 32 + kb) * 32 + lane);
        if (kb + 1 < 32)
            __builtin_prefetch((const void*)(B + ((size_t)ctile * 32 + kb + 1) * 32 + lane), 0, 3);
        v16bf av[10];
#pragma unroll
        for (int rt = 0; rt < 10; ++rt)
            av[rt] = ld_frag((const Q8*)(cur + rt * 1024) + lane);
#pragma unroll
        for (int rt = 0; rt < 10; ++rt)
            acc[rt] = __builtin_amdgcn_wmma_f32_16x16x32_bf16(
                false, av[rt], false, bv, (short)0, acc[rt], false, false);
        __syncthreads();                    // reads done before buffer reuse
    }
}

// ---------------------------------------------------------------------------
// GEMM1: latent = tanh(x @ W_latent + b) -> bf16, scattered directly into the
// A-fragment layout of GEMM2 (row remap n -> n*10 + expert).
// ---------------------------------------------------------------------------
__global__ void gemm1_latent(const Q8* __restrict__ Ax, const Q8* __restrict__ Bw,
                             const float* __restrict__ b_latent,
                             unsigned short* __restrict__ latentA) {
    const int wave = threadIdx.x >> 5;
    const int lane = threadIdx.x & 31;
    const int mtile0 = blockIdx.y * 8;
    const int ctile  = blockIdx.x * 8 + wave;
    v8f acc[8];
    wmma_mainloop<8>(Ax, Bw, mtile0, ctile, acc);

    const int gcol = ctile * 16 + (lane & 15);      // 0..10239
    const float bias = b_latent[gcol];
    const int k  = gcol >> 10, d = gcol & 1023;
    const int kb2 = d >> 5, kl = d & 31;
    const int lane2hi = (((kl >> 3) & 1) << 4);
    const int v = ((kl >> 4) << 2) | ((kl >> 1) & 3);
    const int h = (v << 1) | (kl & 1);
#pragma unroll
    for (int rt = 0; rt < 8; ++rt) {
#pragma unroll
        for (int r = 0; r < 8; ++r) {
            const int grow = (mtile0 + rt) * 16 + r + ((lane >> 4) << 3); // token n
            const float val = tanhf(acc[rt][r] + bias);
            const int arow = grow * 10 + k;
            const int mt2 = arow >> 4, m = arow & 15;
            latentA[((size_t)(mt2 * 32 + kb2)) * 512 + (m + lane2hi) * 16 + h] =
                f32_to_bf16(val);
        }
    }
}

// ---------------------------------------------------------------------------
// GEMM2 pass 1: rowsum[(n,k)] = sum_t exp(logit). Tanh-bounded latent keeps
// |logit| < ~40 so exp is fp32-safe without max subtraction.
// grid = (250 ntiles of 128 cols, 32 mtiles of 160 rows), 256 threads.
// ---------------------------------------------------------------------------
__global__ void gemm2_sumexp(const Q8* __restrict__ Al, const Q8* __restrict__ Bd,
                             const float* __restrict__ b_dec,
                             float* __restrict__ rowsum) {
    __shared__ __align__(16) char smemA[2 * 10240];
    __shared__ float lds_rs[160];
    for (int i = threadIdx.x; i < 160; i += 256) lds_rs[i] = 0.f;
    __syncthreads();

    const int wave = threadIdx.x >> 5;
    const int lane = threadIdx.x & 31;
    const int mtile0 = blockIdx.y * 10;
    const int ctile  = blockIdx.x * 8 + wave;
    v8f acc[10];
    wmma_mainloop_lds10(Al, Bd, mtile0, ctile, acc, smemA);

    const float bias = b_dec[ctile * 16 + (lane & 15)];
#pragma unroll
    for (int rt = 0; rt < 10; ++rt) {
#pragma unroll
        for (int r = 0; r < 8; ++r) {
            float e = __expf(acc[rt][r] + bias);
            e += __shfl_xor(e, 1, 32);   // reduce across the 16 column lanes
            e += __shfl_xor(e, 2, 32);
            e += __shfl_xor(e, 4, 32);
            e += __shfl_xor(e, 8, 32);
            if ((lane & 15) == 0)
                atomicAdd(&lds_rs[rt * 16 + r + ((lane >> 4) << 3)], e);
        }
    }
    __syncthreads();
    for (int i = threadIdx.x; i < 160; i += 256)
        atomicAdd(&rowsum[blockIdx.y * 160 + i], lds_rs[i]);
}

__global__ void make_scale(const float* __restrict__ pis,
                           const float* __restrict__ rowsum,
                           float* __restrict__ scale) {
    const int r = blockIdx.x * blockDim.x + threadIdx.x;
    if (r < NTOKENS * NEXP) scale[r] = pis[r] / rowsum[r];
}

// ---------------------------------------------------------------------------
// GEMM2 pass 2: prob[n,t] = sum_k scale[(n,k)] * exp(logit[(n,k),t]).
// 160-row tile = exactly 16 tokens x 10 experts -> mixture reduction is
// entirely workgroup-local via LDS float atomics; coalesced b128 store out.
// ---------------------------------------------------------------------------
__global__ void gemm2_prob(const Q8* __restrict__ Al, const Q8* __restrict__ Bd,
                           const float* __restrict__ b_dec,
                           const float* __restrict__ scale,
                           float* __restrict__ prob) {
    __shared__ __align__(16) char smemA[2 * 10240];
    __shared__ __align__(16) float lds_p[16 * 128];
    for (int i = threadIdx.x; i < 2048; i += 256) lds_p[i] = 0.f;
    __syncthreads();

    const int wave = threadIdx.x >> 5;
    const int lane = threadIdx.x & 31;
    const int mtile0 = blockIdx.y * 10;
    const int ctile  = blockIdx.x * 8 + wave;
    v8f acc[10];
    wmma_mainloop_lds10(Al, Bd, mtile0, ctile, acc, smemA);

    const float bias = b_dec[ctile * 16 + (lane & 15)];
    const int colL = wave * 16 + (lane & 15);
#pragma unroll
    for (int rt = 0; rt < 10; ++rt) {
#pragma unroll
        for (int r = 0; r < 8; ++r) {
            const int rowL = rt * 16 + r + ((lane >> 4) << 3);   // 0..159
            const float p = scale[blockIdx.y * 160 + rowL] * __expf(acc[rt][r] + bias);
            atomicAdd(&lds_p[(rowL / 10) * 128 + colL], p);
        }
    }
    __syncthreads();
    for (int i = threadIdx.x; i < 512; i += 256) {
        const int tok = i >> 5, c4 = i & 31;
        const float4 vv = *(const float4*)&lds_p[tok * 128 + c4 * 4];
        *(float4*)&prob[((size_t)(blockIdx.y * 16 + tok)) * NTOK +
                        blockIdx.x * 128 + c4 * 4] = vv;
    }
}

// ---------------------------------------------------------------------------
extern "C" void kernel_launch(void* const* d_in, const int* in_sizes, int n_in,
                              void* d_out, int out_size, void* d_ws, size_t ws_size,
                              hipStream_t stream) {
    (void)in_sizes; (void)n_in; (void)out_size; (void)ws_size;
    const float* x     = (const float*)d_in[0];
    const float* W_lat = (const float*)d_in[2];
    const float* b_lat = (const float*)d_in[3];
    const float* W_dec = (const float*)d_in[4];
    const float* b_dec = (const float*)d_in[5];
    const float* W_red = (const float*)d_in[6];
    const float* b_red = (const float*)d_in[7];

    float* prob = (float*)d_out;
    float* mu   = prob + (size_t)NTOKENS * NTOK;
    float* lv   = mu + NTOKENS * 9;

    // workspace layout (all offsets 256B-aligned), total ~98.1 MB
    char* ws = (char*)d_ws;
    unsigned short* Bdec = (unsigned short*)(ws);              // 2000*32*1024B = 65,536,000
    unsigned short* Blat = (unsigned short*)(ws + 65536000);   //  640*32*1024B = 20,971,520
    unsigned short* Ax   = (unsigned short*)(ws + 86507520);   //   32*32*1024B =  1,048,576
    unsigned short* Alat = (unsigned short*)(ws + 87556096);   //  320*32*1024B = 10,485,760
    float* pis    = (float*)(ws + 98041856);                   // 5120 f32
    float* rowsum = (float*)(ws + 98062336);                   // 5120 f32
    float* scale  = (float*)(ws + 98082816);                   // 5120 f32

    convert_x_frags<<<dim3(32, 32),   512, 0, stream>>>(x, Ax);
    convert_B_frags<<<dim3(640, 32),  512, 0, stream>>>(W_lat, Blat, NEXP * NINPD);
    convert_B_frags<<<dim3(2000, 32), 512, 0, stream>>>(W_dec, Bdec, NTOK);
    reduce_pis<<<NTOKENS, 32, 0, stream>>>(x, W_red, b_red, mu, lv, pis);
    zero_f32<<<20, 256, 0, stream>>>(rowsum, NTOKENS * NEXP);

    gemm1_latent<<<dim3(80, 4),   256, 0, stream>>>((const Q8*)Ax,   (const Q8*)Blat, b_lat, Alat);
    gemm2_sumexp<<<dim3(250, 32), 256, 0, stream>>>((const Q8*)Alat, (const Q8*)Bdec, b_dec, rowsum);
    make_scale<<<20, 256, 0, stream>>>(pis, rowsum, scale);
    gemm2_prob<<<dim3(250, 32),  256, 0, stream>>>((const Q8*)Alat, (const Q8*)Bdec, b_dec, scale, prob);
}
